// ConformalLayers_50087908606095
// MI455X (gfx1250) — matrix operand order repaired and checked by reading, not compiled, not run on hardware
//
#include <hip/hip_runtime.h>
#include <hip/hip_bf16.h>
#include <math.h>

// ---------------------------------------------------------------------------
// ConformalLayers on MI455X (gfx1250, wave32, WMMA + async-to-LDS).
//   out[b, o] = (M @ x_b)[o] / ( s*||x_b|| + x_b^T T x_b )
// ~490 GFLOP vs ~0.56 GB -> compute bound. fp32 -> bf16 (RNE) once, both
// GEMMs via v_wmma_f32_16x16x32_bf16 (f32 accumulate). bf16 operands
// (~125 MB) are L2-resident (192 MB L2). Main GEMM: double-buffered LDS
// pipeline fed by global_load_async_to_lds_b128 (ASYNCcnt), NT stores for
// the streamed 268 MB output.
// ---------------------------------------------------------------------------

#define INN  3072
#define OUTN 16384
#define NB   4096

typedef __attribute__((ext_vector_type(16))) __bf16 v16bf;
typedef __attribute__((ext_vector_type(8)))  float  v8f;
typedef __attribute__((ext_vector_type(4)))  float  v4f;

union FragB16 { v16bf v; uint4 q[2]; };   // 16 bf16 = 32 B = two b128 chunks

__device__ __forceinline__ unsigned short f2bf(float f) {
  unsigned int u = __float_as_uint(f);
  return (unsigned short)((u + 0x7FFFu + ((u >> 16) & 1u)) >> 16); // RNE
}
__device__ __forceinline__ float bf2f(unsigned short h) {
  return __uint_as_float(((unsigned int)h) << 16);
}

// One async b128 copy: LDS[lds_off] = *(b128*)(base_bytes + goff).  GVS mode:
// uniform SGPR64 base + per-lane unsigned 32-bit byte offset.
__device__ __forceinline__ void async_b128(unsigned lds_off, unsigned goff,
                                           const unsigned short* base) {
  asm volatile("global_load_async_to_lds_b128 %0, %1, %2"
               :: "v"(lds_off), "v"(goff), "s"(base) : "memory");
}

// --- K1: x (f32, batch-major) -> bf16, plus per-batch L2 norm --------------
__global__ void __launch_bounds__(256) k_conv_x(
    const float* __restrict__ x, unsigned short* __restrict__ xbf,
    float* __restrict__ norm) {
  const int b = blockIdx.x;
  const float* xr = x + (size_t)b * INN;
  unsigned short* xw = xbf + (size_t)b * INN;
  float s = 0.f;
  for (int i = threadIdx.x; i < INN; i += 256) {
    float v = xr[i];
    s += v * v;
    xw[i] = f2bf(v);
  }
  for (int off = 16; off > 0; off >>= 1) s += __shfl_down(s, off, 32);
  __shared__ float wsum[8];
  const int lane = threadIdx.x & 31, w = threadIdx.x >> 5;
  if (lane == 0) wsum[w] = s;
  __syncthreads();
  if (threadIdx.x == 0) {
    float t = 0.f;
    #pragma unroll
    for (int i = 0; i < 8; ++i) t += wsum[i];
    norm[b] = sqrtf(t);
  }
}

// --- K2: generic f32 -> bf16 (n multiple of 4) -----------------------------
__global__ void __launch_bounds__(256) k_conv(
    const float* __restrict__ in, unsigned short* __restrict__ out, int n) {
  int i = (blockIdx.x * 256 + threadIdx.x) * 4;
  if (i < n) {
    float4 v = *(const float4*)(in + i);
    ushort4 o;
    o.x = f2bf(v.x); o.y = f2bf(v.y); o.z = f2bf(v.z); o.w = f2bf(v.w);
    *(ushort4*)(out + i) = o;
  }
}

// --- K3: dinv[b] = 1 / (s*||x_b|| + x_b^T T x_b) via WMMA ------------------
__global__ void __launch_bounds__(256) k_quad(
    const unsigned short* __restrict__ Tbf, const unsigned short* __restrict__ Xbf,
    const float* __restrict__ norm, const float* __restrict__ s_ptr,
    float* __restrict__ dinv) {
  const int b0   = blockIdx.x * 16;
  const int lane = threadIdx.x & 31;
  const int wave = threadIdx.x >> 5;
  const int r    = lane & 15;
  const int h    = lane >> 4;

  __shared__ float red[16];
  if (threadIdx.x < 16) red[threadIdx.x] = 0.f;
  __syncthreads();

  const unsigned short* xcol = Xbf + (size_t)(b0 + r) * INN;
  float partial = 0.f;

  for (int mt = wave * 16; mt < INN; mt += 8 * 16) {
    const unsigned short* arow = Tbf + (size_t)(mt + r) * INN;
    v8f acc = {};
    for (int k = 0; k < INN; k += 32) {
      FragB16 a, bx;
      a.q[0] = *(const uint4*)(arow + k + h * 8);
      a.q[1] = *(const uint4*)(arow + k + 16 + h * 8);
      bx.q[0] = *(const uint4*)(xcol + k + h * 8);
      bx.q[1] = *(const uint4*)(xcol + k + 16 + h * 8);
      acc = __builtin_amdgcn_wmma_f32_16x16x32_bf16(
          false, a.v, false, bx.v, (short)0, acc, false, false);
    }
    #pragma unroll
    for (int i = 0; i < 8; ++i) {
      int m = mt + 8 * h + i;
      partial += acc[i] * bf2f(xcol[m]);
    }
  }
  atomicAdd(&red[r], partial);   // ds_add_f32
  __syncthreads();
  if (threadIdx.x < 16) {
    int b = b0 + threadIdx.x;
    dinv[b] = 1.0f / (s_ptr[0] * norm[b] + red[threadIdx.x]);
  }
}

// --- K4: out[b][o] = (M @ X)[o][b] * dinv[b] -------------------------------
// WG tile 128(M) x 128(N) x 32(K); async double-buffered LDS pipeline.
// LDS layout per buffer: A 128 rows x 32 bf16, row stride 80 B (bank spread);
// B 128 cols x 32 bf16, same. Buffer = 2*10240 B; two buffers = 40960 B.
#define ROWB  80
#define MATB  (128 * ROWB)       // 10240
#define BUFB  (2 * MATB)         // 20480

__global__ void __launch_bounds__(256) k_gemm(
    const unsigned short* __restrict__ Mbf, const unsigned short* __restrict__ Xbf,
    const float* __restrict__ dinv, float* __restrict__ out) {
  extern __shared__ char smem[];
  const int tid  = threadIdx.x;
  const int lane = tid & 31, wave = tid >> 5;
  const int r = lane & 15, h = lane >> 4;
  const int wm = wave & 1, wn = wave >> 1;
  const unsigned Mblk = blockIdx.y * 128;
  const unsigned Nblk = blockIdx.x * 128;

  // Each thread stages 2 A-chunks + 2 B-chunks (16 B each) per buffer.
  // Chunk id c in [0,512): row = c>>2, chunk-in-row cp = c&3.
  const int c0 = tid * 2;

  auto stage = [&](int buf, unsigned k) {
    const unsigned lbase = (unsigned)buf * BUFB;
    #pragma unroll
    for (int j = 0; j < 2; ++j) {
      const unsigned c = c0 + j;
      const unsigned row = c >> 2, cp = c & 3;
      const unsigned lds = lbase + row * ROWB + cp * 16;
      async_b128(lds,          (((Mblk + row) * INN + k + cp * 8) * 2u), Mbf);
      async_b128(lds + MATB,   (((Nblk + row) * INN + k + cp * 8) * 2u), Xbf);
    }
  };

  v8f acc[4][2] = {};
  stage(0, 0);

  for (int kt = 0; kt < INN / 32; ++kt) {
    const int cur = kt & 1;
    if (kt + 1 < INN / 32) {
      stage(cur ^ 1, (unsigned)(kt + 1) * 32);
      asm volatile("s_wait_asynccnt 0x4" ::: "memory"); // cur buffer landed
    } else {
      asm volatile("s_wait_asynccnt 0x0" ::: "memory");
    }
    __syncthreads();                                    // all waves staged cur

    const char* sa = smem + cur * BUFB;
    const char* sb = sa + MATB;
    FragB16 a[4], b[2];
    #pragma unroll
    for (int mi = 0; mi < 4; ++mi) {
      const char* p = sa + (wm * 64 + mi * 16 + r) * ROWB;
      a[mi].q[0] = *(const uint4*)(p + h * 16);         // K[h*8 .. +7]
      a[mi].q[1] = *(const uint4*)(p + 32 + h * 16);    // K[16+h*8 .. +7]
    }
    #pragma unroll
    for (int ni = 0; ni < 2; ++ni) {
      const char* p = sb + (wn * 32 + ni * 16 + r) * ROWB;
      b[ni].q[0] = *(const uint4*)(p + h * 16);         // K[h*8 .. +7]
      b[ni].q[1] = *(const uint4*)(p + 32 + h * 16);    // K[16+h*8 .. +7]
    }
    #pragma unroll
    for (int mi = 0; mi < 4; ++mi)
      #pragma unroll
      for (int ni = 0; ni < 2; ++ni)
        acc[mi][ni] = __builtin_amdgcn_wmma_f32_16x16x32_bf16(
            false, a[mi].v, false, b[ni].v, (short)0, acc[mi][ni], false, false);

    __syncthreads();   // reads of cur done before it is overwritten next iter
  }

  // Epilogue: o = Mblk + wm*64 + mi*16 + 8*h + i (contiguous i), b = column.
  // Output is streamed once -> non-temporal stores (don't evict L2 operands).
  #pragma unroll
  for (int ni = 0; ni < 2; ++ni) {
    const int b = Nblk + wn * 32 + ni * 16 + r;
    const float di = dinv[b];
    #pragma unroll
    for (int mi = 0; mi < 4; ++mi) {
      float* p = out + (size_t)b * OUTN + (Mblk + wm * 64 + mi * 16 + 8 * h);
      v4f lo = { acc[mi][ni][0] * di, acc[mi][ni][1] * di,
                 acc[mi][ni][2] * di, acc[mi][ni][3] * di };
      v4f hi = { acc[mi][ni][4] * di, acc[mi][ni][5] * di,
                 acc[mi][ni][6] * di, acc[mi][ni][7] * di };
      __builtin_nontemporal_store(lo, (v4f*)p);
      __builtin_nontemporal_store(hi, (v4f*)(p + 4));
    }
  }
}

// ---------------------------------------------------------------------------
extern "C" void kernel_launch(void* const* d_in, const int* in_sizes, int n_in,
                              void* d_out, int out_size, void* d_ws, size_t ws_size,
                              hipStream_t stream) {
  const float* x  = (const float*)d_in[0];   // [4096, 3072]
  const float* M  = (const float*)d_in[1];   // [16384, 3072]
  const float* s  = (const float*)d_in[2];   // [1]
  const float* T  = (const float*)d_in[3];   // [3072, 3072]
  float* out = (float*)d_out;                // [4096, 16384]

  char* w = (char*)d_ws;
  const size_t szM = (size_t)OUTN * INN * sizeof(unsigned short); // 100663296
  const size_t szX = (size_t)NB   * INN * sizeof(unsigned short); //  25165824
  const size_t szT = (size_t)INN  * INN * sizeof(unsigned short); //  18874368
  unsigned short* Mbf  = (unsigned short*)(w);
  unsigned short* Xbf  = (unsigned short*)(w + szM);
  unsigned short* Tbf  = (unsigned short*)(w + szM + szX);
  float*          norm = (float*)(w + szM + szX + szT);
  float*          dinv = (float*)(w + szM + szX + szT + NB * sizeof(float));

  k_conv_x<<<NB, 256, 0, stream>>>(x, Xbf, norm);
  {
    int nM = OUTN * INN;
    k_conv<<<(nM / 4 + 255) / 256, 256, 0, stream>>>(M, Mbf, nM);
    int nT = INN * INN;
    k_conv<<<(nT / 4 + 255) / 256, 256, 0, stream>>>(T, Tbf, nT);
  }
  k_quad<<<NB / 16, 256, 0, stream>>>(Tbf, Xbf, norm, s, dinv);
  k_gemm<<<dim3(NB / 128, OUTN / 128), 256, 40960, stream>>>(Mbf, Xbf, dinv, out);
}
